// EntityAttentionRNNAgent_MI_72911364816897
// MI455X (gfx1250) — matrix operand order repaired and checked by reading, not compiled, not run on hardware
//
#include <hip/hip_runtime.h>
#include <hip/hip_bf16.h>

// EntityAttentionRNNAgent for MI455X (gfx1250, wave32, WMMA).
// All large GEMMs use v_wmma_f32_16x16x32_f16 (f32 accumulate).
// Weights are pre-converted to f16 once so the hot loops contain no
// f32->f16 conversion VALU work (only b128 loads + WMMA).
// GEMM uses double-buffered LDS tiles (one barrier per K-step).
// Shapes: BS=32 TS=64 NE=64 ED=128 AE=256 NH=4 HD=64 RH=256 NA=8 NACT=64.

typedef __attribute__((ext_vector_type(16))) _Float16 v16h;
typedef __attribute__((ext_vector_type(8)))  _Float16 h8;
typedef __attribute__((ext_vector_type(8)))  float    v8f;

#define NEG_BIG (-10000000000.0f)

// ---------------------------------------------------------------------------
// Bulk f32 -> f16 conversion (n must be a multiple of 2048; launch n/2048 blk)
// ---------------------------------------------------------------------------
__global__ __launch_bounds__(256) void cvt_f32_f16(
    const float* __restrict__ src, _Float16* __restrict__ dst)
{
  size_t i = ((size_t)blockIdx.x * 256 + threadIdx.x) * 8;
  h8 hv;
#pragma unroll
  for (int k = 0; k < 8; ++k) hv[k] = (_Float16)src[i + k];
  *(h8*)(dst + i) = hv;
}

// ---------------------------------------------------------------------------
// Generic GEMM: C[M,N] = act(A[M,K] @ Wh[N,K]^T + bias)
// A: f32 or f16 row-major; Wh: f16 row-major (N,K); C: f32 or f16.
// Block = 256 threads = 8 waves (4 M-waves x 2 N-waves).
// Tile: 128 (M) x 32*NPW (N); each wave: 2 M-tiles x NPW N-tiles
// -> 2*NPW WMMAs per 32-wide K-step. Double-buffered LDS staging:
// stage K-step k+1 while computing k; one __syncthreads per step.
// Requires M%128==0, N%(32*NPW)==0, K%32==0 (all call sites satisfy this).
// ---------------------------------------------------------------------------
template <typename TA, typename TC, int NPW, bool RELU>
__global__ __launch_bounds__(256) void gemm_wmma(
    const TA* __restrict__ A, const _Float16* __restrict__ Wh,
    const float* __restrict__ bias, TC* __restrict__ C,
    int M, int N, int K)
{
  constexpr int NT  = 32 * NPW;
  constexpr int ASZ = 128 * 40;
  constexpr int BSZ = NT * 40;
  __shared__ __align__(16) _Float16 As[2 * ASZ];
  __shared__ __align__(16) _Float16 Bs[2 * BSZ];

  const int tid    = threadIdx.x;
  const int wave   = tid >> 5;
  const int lane   = tid & 31;
  const int wm     = wave >> 1;     // 0..3 -> 32-row group (2 tiles)
  const int wn     = wave & 1;      // 0..1 -> NPW N-tiles each
  const int m0     = blockIdx.x * 128;
  const int n0     = blockIdx.y * NT;
  const int half16 = lane >> 4;
  const int l15    = lane & 15;

  v8f acc[2][NPW];
#pragma unroll
  for (int mi = 0; mi < 2; ++mi)
#pragma unroll
    for (int j = 0; j < NPW; ++j)
#pragma unroll
      for (int i = 0; i < 8; ++i) acc[mi][j][i] = 0.0f;

  auto stage = [&](int buf, int k0) {
#pragma unroll
    for (int it = 0; it < 2; ++it) {              // A tile (128 x 32)
      int idx  = tid + it * 256;
      int arow = idx >> 2;
      int ak   = (idx & 3) * 8;
      const TA* src = A + (size_t)(m0 + arow) * K + (k0 + ak);
      h8 hv;
      if constexpr (sizeof(TA) == 4) {
#pragma unroll
        for (int i = 0; i < 8; ++i) hv[i] = (_Float16)((const float*)src)[i];
      } else {
        hv = *(const h8*)src;
      }
      *(h8*)&As[buf * ASZ + arow * 40 + ak] = hv;
    }
#pragma unroll
    for (int it = 0; it < (NT * 4) / 256; ++it) { // B tile (NT x 32)
      int idx = tid + it * 256;
      int br  = idx >> 2;
      int bk  = (idx & 3) * 8;
      *(h8*)&Bs[buf * BSZ + br * 40 + bk] =
          *(const h8*)(Wh + (size_t)(n0 + br) * K + (k0 + bk));
    }
  };

  stage(0, 0);
  __syncthreads();

  const int ksteps = K >> 5;
  for (int ks = 0; ks < ksteps; ++ks) {
    const int cur = ks & 1;
    if (ks + 1 < ksteps) stage(cur ^ 1, (ks + 1) << 5);

    // A fragments: row = l15, 16 contiguous K at half16*16 (matches B).
    v16h afrag[2];
#pragma unroll
    for (int mi = 0; mi < 2; ++mi) {
      const _Float16* ap =
          &As[cur * ASZ + (wm * 32 + mi * 16 + l15) * 40 + half16 * 16];
      h8 lo = *(const h8*)ap;
      h8 hi = *(const h8*)(ap + 8);
#pragma unroll
      for (int i = 0; i < 8; ++i) { afrag[mi][i] = lo[i]; afrag[mi][i + 8] = hi[i]; }
    }
#pragma unroll
    for (int j = 0; j < NPW; ++j) {
      // B fragment: col = l15, 16 contiguous K at half16*16 (B = Wh row).
      const _Float16* bp =
          &Bs[cur * BSZ + ((wn * NPW + j) * 16 + l15) * 40 + half16 * 16];
      h8 lo = *(const h8*)bp;
      h8 hi = *(const h8*)(bp + 8);
      v16h bfrag;
#pragma unroll
      for (int i = 0; i < 8; ++i) { bfrag[i] = lo[i]; bfrag[i + 8] = hi[i]; }
#pragma unroll
      for (int mi = 0; mi < 2; ++mi)
        acc[mi][j] = __builtin_amdgcn_wmma_f32_16x16x32_f16(
            false, afrag[mi], false, bfrag, (short)0, acc[mi][j], false, false);
    }
    __syncthreads();  // staging(cur^1) done + reads(cur) done
  }

  // Epilogue (C/D layout: col = l15, VGPR r = row r + 8*half16)
  const int rbase = half16 * 8;
#pragma unroll
  for (int mi = 0; mi < 2; ++mi)
#pragma unroll
    for (int j = 0; j < NPW; ++j) {
      int col  = n0 + (wn * NPW + j) * 16 + l15;
      float bv = bias ? bias[col] : 0.0f;
#pragma unroll
      for (int r = 0; r < 8; ++r) {
        int row = m0 + wm * 32 + mi * 16 + rbase + r;
        float v = acc[mi][j][r] + bv;
        if (RELU) v = v > 0.0f ? v : 0.0f;
        C[(size_t)row * N + col] = (TC)v;
      }
    }
}

// ---------------------------------------------------------------------------
// Fused GRU step: gh = h(t-1) @ Whh^T + bhh via WMMA, then gate math.
// Grid = 16 blocks x 16 rows of (BS*NA=256). Each of 8 waves owns 6 N-tiles
// (8*6*16 = 768 columns). h(t-1) read from hs (or h0 at t==0); h(t) -> hs.
// gi is precomputed (incl. bih), layout (BS,TS,NA,768). whh pre-cvt f16.
// ---------------------------------------------------------------------------
__global__ __launch_bounds__(256) void gru_step_kernel(
    const float* __restrict__ gi, const _Float16* __restrict__ whh,
    const float* __restrict__ bhh, const float* __restrict__ h0,
    float* __restrict__ hs, int t)
{
  __shared__ __align__(16) _Float16 Hs[16 * 264];
  __shared__ float Gh[16 * 776];

  const int tid    = threadIdx.x;
  const int wave   = tid >> 5;
  const int lane   = tid & 31;
  const int half16 = lane >> 4;
  const int l15    = lane & 15;
  const int rowbase = blockIdx.x * 16;

  float hprev[16];
#pragma unroll
  for (int i = 0; i < 16; ++i) {
    int idx = tid + i * 256;
    int r = idx >> 8, c = idx & 255;
    int gr = rowbase + r;
    int bs = gr >> 3, a = gr & 7;
    float hv = (t == 0) ? h0[(size_t)gr * 256 + c]
                        : hs[((size_t)(bs * 64 + (t - 1)) * 8 + a) * 256 + c];
    hprev[i] = hv;
    Hs[r * 264 + c] = (_Float16)hv;
  }
  __syncthreads();

  v8f acc[6];
#pragma unroll
  for (int j = 0; j < 6; ++j)
#pragma unroll
    for (int i = 0; i < 8; ++i) acc[j][i] = 0.0f;

#pragma unroll
  for (int k0 = 0; k0 < 256; k0 += 32) {
    v16h afrag;
    {
      const _Float16* ap = &Hs[l15 * 264 + k0 + half16 * 16];
      h8 lo = *(const h8*)ap;
      h8 hi = *(const h8*)(ap + 8);
#pragma unroll
      for (int i = 0; i < 8; ++i) { afrag[i] = lo[i]; afrag[i + 8] = hi[i]; }
    }
#pragma unroll
    for (int j = 0; j < 6; ++j) {
      int n = (wave * 6 + j) * 16 + l15;
      const _Float16* wr = whh + (size_t)n * 256 + k0 + half16 * 16;
      h8 lo = *(const h8*)wr;      // whh f16, L2-resident (393 KB)
      h8 hi = *(const h8*)(wr + 8);
      v16h bfrag;
#pragma unroll
      for (int i = 0; i < 8; ++i) { bfrag[i] = lo[i]; bfrag[i + 8] = hi[i]; }
      acc[j] = __builtin_amdgcn_wmma_f32_16x16x32_f16(
          false, afrag, false, bfrag, (short)0, acc[j], false, false);
    }
  }

#pragma unroll
  for (int j = 0; j < 6; ++j) {
    int ncol = (wave * 6 + j) * 16 + l15;
    float bv = bhh[ncol];
#pragma unroll
    for (int r = 0; r < 8; ++r)
      Gh[(half16 * 8 + r) * 776 + ncol] = acc[j][r] + bv;
  }
  __syncthreads();

#pragma unroll
  for (int i = 0; i < 16; ++i) {
    int idx = tid + i * 256;
    int r = idx >> 8, c = idx & 255;
    int gr = rowbase + r;
    int bs = gr >> 3, a = gr & 7;
    size_t gib = ((size_t)(bs * 64 + t) * 8 + a) * 768;
    float ir  = gi[gib + c];
    float iz  = gi[gib + 256 + c];
    float inn = gi[gib + 512 + c];
    float hr = Gh[r * 776 + c];
    float hz = Gh[r * 776 + 256 + c];
    float hn = Gh[r * 776 + 512 + c];
    float rg = 1.0f / (1.0f + __expf(-(ir + hr)));
    float zg = 1.0f / (1.0f + __expf(-(iz + hz)));
    float ng = tanhf(inn + rg * hn);
    float hnew = (1.0f - zg) * ng + zg * hprev[i];
    hs[((size_t)(bs * 64 + t) * 8 + a) * 256 + c] = hnew;
  }
}

// ---------------------------------------------------------------------------
// Attention core: one 64-thread block per (b,h,a). logits->softmax->ctx.
// q: (B*NA,256) f16; kv: (B*NE,512) f16 (k cols 0..255, v cols 256..511).
// ---------------------------------------------------------------------------
__global__ __launch_bounds__(64) void attn_kernel(
    const _Float16* __restrict__ qb, const _Float16* __restrict__ kv,
    const int* __restrict__ obs_mask, _Float16* __restrict__ ctx)
{
  const int tid = threadIdx.x;           // 0..63
  const int bid = blockIdx.x;            // b*32 + h*8 + a
  const int a = bid & 7;
  const int h = (bid >> 3) & 3;
  const int b = bid >> 5;

  __shared__ float qs[64];
  __shared__ float ls[64];

  qs[tid] = (float)qb[((size_t)(b * 8 + a) * 256) + h * 64 + tid];
  __syncthreads();

  const _Float16* kp = kv + ((size_t)(b * 64 + tid) * 512) + h * 64;
  float acc = 0.0f;
#pragma unroll 8
  for (int d = 0; d < 64; ++d) acc += qs[d] * (float)kp[d];
  int m = obs_mask[((size_t)b * 64 + a) * 64 + tid];
  ls[tid] = (m != 0) ? -__builtin_inff() : acc * 0.125f;  // 1/sqrt(64)
  __syncthreads();

  float mx = -__builtin_inff();
  for (int e = 0; e < 64; ++e) mx = ls[e] > mx ? ls[e] : mx;
  bool dead = (mx == -__builtin_inff());  // all keys masked
  float wt = dead ? 0.0f : __expf(ls[tid] - mx);
  __syncthreads();
  ls[tid] = wt;
  __syncthreads();
  float sum = 0.0f;
  for (int e = 0; e < 64; ++e) sum += ls[e];
  float inv = dead ? 0.0f : 1.0f / sum;   // dead => ctx = 0 (matches ref)

  float o = 0.0f;
  for (int e = 0; e < 64; ++e)
    o += (ls[e] * inv) * (float)kv[((size_t)(b * 64 + e) * 512) + 256 + h * 64 + tid];
  ctx[((size_t)(b * 8 + a) * 256) + h * 64 + tid] = (_Float16)o;
}

// ---------------------------------------------------------------------------
// Small data-movement / epilogue kernels
// ---------------------------------------------------------------------------
// Gather first-NA entity rows of x1 (f16, 256 cols): (B,64,256)->(B,8,256)
__global__ __launch_bounds__(256) void gather_agents_h8(
    const _Float16* __restrict__ src, _Float16* __restrict__ dst)
{
  int idx = blockIdx.x * 256 + threadIdx.x;   // 16384*256/8 = 524288
  int row = idx >> 5;
  int c   = (idx & 31) * 8;
  int b = row >> 3, a = row & 7;
  *(h8*)(dst + (size_t)row * 256 + c) =
      *(const h8*)(src + ((size_t)(b * 64 + a) * 256) + c);
}

// Gather agent entity rows, f32 -> f16: (B,64,128)->(B,8,128)
__global__ __launch_bounds__(256) void gather_ent_f32_h8(
    const float* __restrict__ ent, _Float16* __restrict__ dst)
{
  int idx = blockIdx.x * 256 + threadIdx.x;   // 16384*128/8 = 262144
  int row = idx >> 4;
  int c   = (idx & 15) * 8;
  const float* s = ent + ((size_t)((row >> 3) * 64 + (row & 7)) * 128) + c;
  h8 hv;
#pragma unroll
  for (int i = 0; i < 8; ++i) hv[i] = (_Float16)s[i];
  *(h8*)(dst + (size_t)row * 128 + c) = hv;
}

// Zero f16 rows (256 cols) where agent mask is set
__global__ __launch_bounds__(256) void mask_zero_h8(
    _Float16* __restrict__ x, const int* __restrict__ em)
{
  int idx = blockIdx.x * 256 + threadIdx.x;   // 524288
  int row = idx >> 5;
  int c   = (idx & 31) * 8;
  if (em[(row >> 3) * 64 + (row & 7)] != 0) {
    h8 z;
#pragma unroll
    for (int i = 0; i < 8; ++i) z[i] = (_Float16)0.0f;
    *(h8*)(x + (size_t)row * 256 + c) = z;
  }
}

// Overwrite f32 head outputs (64 cols) with `val` where agent mask is set
__global__ __launch_bounds__(256) void mask_head(
    float* __restrict__ o, const int* __restrict__ em, float val)
{
  int idx = blockIdx.x * 256 + threadIdx.x;   // 16384*64 = 1048576
  int row = idx >> 6;
  if (em[(row >> 3) * 64 + (row & 7)] != 0) o[idx] = val;
}

// ---------------------------------------------------------------------------
extern "C" void kernel_launch(void* const* d_in, const int* in_sizes, int n_in,
                              void* d_out, int out_size, void* d_ws, size_t ws_size,
                              hipStream_t stream) {
  (void)in_sizes; (void)n_in; (void)out_size; (void)ws_size;
  const float* entities   = (const float*)d_in[0];
  const int*   obs_mask   = (const int*)d_in[1];
  const int*   entity_mask= (const int*)d_in[2];
  const float* hidden_q   = (const float*)d_in[3];
  const float* hidden_pi  = (const float*)d_in[4];
  const float* q_fc1_w  = (const float*)d_in[5];
  const float* q_fc1_b  = (const float*)d_in[6];
  const float* q_fc2_w  = (const float*)d_in[7];
  const float* q_fc2_b  = (const float*)d_in[8];
  const float* q_gru_wih= (const float*)d_in[9];
  const float* q_gru_whh= (const float*)d_in[10];
  const float* q_gru_bih= (const float*)d_in[11];
  const float* q_gru_bhh= (const float*)d_in[12];
  const float* q_fcq_w  = (const float*)d_in[13];
  const float* q_fcq_b  = (const float*)d_in[14];
  const float* q_fcpi_w = (const float*)d_in[15];
  const float* q_fcpi_b = (const float*)d_in[16];
  const float* p_fc1_w  = (const float*)d_in[17];
  const float* p_fc1_b  = (const float*)d_in[18];
  const float* p_fc2_w  = (const float*)d_in[19];
  const float* p_fc2_b  = (const float*)d_in[20];
  const float* p_gru_wih= (const float*)d_in[21];
  const float* p_gru_whh= (const float*)d_in[22];
  const float* p_gru_bih= (const float*)d_in[23];
  const float* p_gru_bhh= (const float*)d_in[24];
  const float* p_fcpi_w = (const float*)d_in[27];
  const float* p_fcpi_b = (const float*)d_in[28];
  const float* q_attn_in_w  = (const float*)d_in[29];
  const float* q_attn_out_w = (const float*)d_in[30];
  const float* q_attn_out_b = (const float*)d_in[31];

  // Workspace layout (~267 MB; gi buffers alias the dead kv region).
  char* ws = (char*)d_ws;
  size_t off = 0;
  auto alloc = [&](size_t bytes) -> void* {
    void* p = ws + off;
    off += (bytes + 255) & ~(size_t)255;
    return p;
  };
  _Float16* x1   = (_Float16*)alloc(131072ull * 256 * 2); // relu(fc1) all ents
  _Float16* kvb  = (_Float16*)alloc(131072ull * 512 * 2); // k|v
  _Float16* xq   = (_Float16*)alloc(16384ull * 256 * 2);  // x1 agent rows
  _Float16* qb   = (_Float16*)alloc(16384ull * 256 * 2);  // q projections
  _Float16* ctx  = (_Float16*)alloc(16384ull * 256 * 2);  // attention out
  _Float16* x2   = (_Float16*)alloc(16384ull * 256 * 2);
  _Float16* x3q  = (_Float16*)alloc(16384ull * 256 * 2);
  _Float16* entag= (_Float16*)alloc(16384ull * 128 * 2);
  _Float16* x2p  = (_Float16*)alloc(16384ull * 256 * 2);
  _Float16* x3p  = (_Float16*)alloc(16384ull * 256 * 2);
  // Pre-converted f16 weights
  _Float16* wf_q_fc1   = (_Float16*)alloc(256ull * 128 * 2);
  _Float16* wf_q_ain   = (_Float16*)alloc(768ull * 256 * 2);
  _Float16* wf_q_aout  = (_Float16*)alloc(256ull * 256 * 2);
  _Float16* wf_q_fc2   = (_Float16*)alloc(256ull * 256 * 2);
  _Float16* wf_q_wih   = (_Float16*)alloc(768ull * 256 * 2);
  _Float16* wf_q_whh   = (_Float16*)alloc(768ull * 256 * 2);
  _Float16* wf_q_fcq   = (_Float16*)alloc(64ull * 256 * 2);
  _Float16* wf_q_fcpi  = (_Float16*)alloc(64ull * 256 * 2);
  _Float16* wf_p_fc1   = (_Float16*)alloc(256ull * 128 * 2);
  _Float16* wf_p_fc2   = (_Float16*)alloc(256ull * 256 * 2);
  _Float16* wf_p_wih   = (_Float16*)alloc(768ull * 256 * 2);
  _Float16* wf_p_whh   = (_Float16*)alloc(768ull * 256 * 2);
  _Float16* wf_p_fcpi  = (_Float16*)alloc(64ull * 256 * 2);
  // gi buffers reuse kv region (kv dead after attention; gi computed after)
  float* giq = (float*)kvb;
  float* gip = (float*)((char*)kvb + 16384ull * 768 * 4);

  float* out   = (float*)d_out;
  float* oq    = out;                       // (BS,TS,NA,64)
  float* opi   = out + 1048576;             // (BS,TS,NA,64)
  float* piavg = out + 2097152;             // (BS,TS,NA,64)
  float* hq    = out + 3145728;             // (BS,TS,NA,256)
  float* hpi   = out + 3145728 + 4194304;   // (BS,TS,NA,256)

  dim3 blk(256);
  auto cvt = [&](const float* s, _Float16* d, size_t n) {
    cvt_f32_f16<<<(unsigned)(n / 2048), blk, 0, stream>>>(s, d);
  };

  // ---- weight pre-conversion (once, ~2.6 MB total) ----
  cvt(q_fc1_w,      wf_q_fc1,  256 * 128);
  cvt(q_attn_in_w,  wf_q_ain,  768 * 256);
  cvt(q_attn_out_w, wf_q_aout, 256 * 256);
  cvt(q_fc2_w,      wf_q_fc2,  256 * 256);
  cvt(q_gru_wih,    wf_q_wih,  768 * 256);
  cvt(q_gru_whh,    wf_q_whh,  768 * 256);
  cvt(q_fcq_w,      wf_q_fcq,  64 * 256);
  cvt(q_fcpi_w,     wf_q_fcpi, 64 * 256);
  cvt(p_fc1_w,      wf_p_fc1,  256 * 128);
  cvt(p_fc2_w,      wf_p_fc2,  256 * 256);
  cvt(p_gru_wih,    wf_p_wih,  768 * 256);
  cvt(p_gru_whh,    wf_p_whh,  768 * 256);
  cvt(p_fcpi_w,     wf_p_fcpi, 64 * 256);

  // ---- q module (with attention) ----
  gemm_wmma<float, _Float16, 4, true><<<dim3(1024, 2), blk, 0, stream>>>(
      entities, wf_q_fc1, q_fc1_b, x1, 131072, 256, 128);
  gemm_wmma<_Float16, _Float16, 4, false><<<dim3(1024, 4), blk, 0, stream>>>(
      x1, wf_q_ain + 256 * 256, nullptr, kvb, 131072, 512, 256);
  gather_agents_h8<<<2048, blk, 0, stream>>>(x1, xq);
  gemm_wmma<_Float16, _Float16, 4, false><<<dim3(128, 2), blk, 0, stream>>>(
      xq, wf_q_ain, nullptr, qb, 16384, 256, 256);
  attn_kernel<<<65536, 64, 0, stream>>>(qb, kvb, obs_mask, ctx);
  gemm_wmma<_Float16, _Float16, 4, false><<<dim3(128, 2), blk, 0, stream>>>(
      ctx, wf_q_aout, q_attn_out_b, x2, 16384, 256, 256);
  mask_zero_h8<<<2048, blk, 0, stream>>>(x2, entity_mask);
  gemm_wmma<_Float16, _Float16, 4, true><<<dim3(128, 2), blk, 0, stream>>>(
      x2, wf_q_fc2, q_fc2_b, x3q, 16384, 256, 256);
  gemm_wmma<_Float16, float, 4, false><<<dim3(128, 6), blk, 0, stream>>>(
      x3q, wf_q_wih, q_gru_bih, giq, 16384, 768, 256);
  for (int t = 0; t < 64; ++t)
    gru_step_kernel<<<16, blk, 0, stream>>>(giq, wf_q_whh, q_gru_bhh,
                                            hidden_q, hq, t);
  gemm_wmma<float, float, 2, false><<<dim3(128, 1), blk, 0, stream>>>(
      hq, wf_q_fcq, q_fcq_b, oq, 16384, 64, 256);
  gemm_wmma<float, float, 2, false><<<dim3(128, 1), blk, 0, stream>>>(
      hq, wf_q_fcpi, q_fcpi_b, opi, 16384, 64, 256);
  mask_head<<<4096, blk, 0, stream>>>(oq, entity_mask, 0.0f);
  mask_head<<<4096, blk, 0, stream>>>(opi, entity_mask, NEG_BIG);

  // ---- p module (no attention; only agent entities ever used) ----
  gather_ent_f32_h8<<<1024, blk, 0, stream>>>(entities, entag);
  gemm_wmma<_Float16, _Float16, 4, true><<<dim3(128, 2), blk, 0, stream>>>(
      entag, wf_p_fc1, p_fc1_b, x2p, 16384, 256, 128);
  gemm_wmma<_Float16, _Float16, 4, true><<<dim3(128, 2), blk, 0, stream>>>(
      x2p, wf_p_fc2, p_fc2_b, x3p, 16384, 256, 256);
  gemm_wmma<_Float16, float, 4, false><<<dim3(128, 6), blk, 0, stream>>>(
      x3p, wf_p_wih, p_gru_bih, gip, 16384, 768, 256);
  for (int t = 0; t < 64; ++t)
    gru_step_kernel<<<16, blk, 0, stream>>>(gip, wf_p_whh, p_gru_bhh,
                                            hidden_pi, hpi, t);
  gemm_wmma<float, float, 2, false><<<dim3(128, 1), blk, 0, stream>>>(
      hpi, wf_p_fcpi, p_fcpi_b, piavg, 16384, 64, 256);
  mask_head<<<4096, blk, 0, stream>>>(piavg, entity_mask, NEG_BIG);
}
